// QnnFormerAttention_61512521613698
// MI455X (gfx1250) — matrix operand order repaired and checked by reading, not compiled, not run on hardware
//
#include <hip/hip_runtime.h>
#include <hip/hip_bf16.h>
#include <math.h>

// -------- CDNA5 WMMA types --------
typedef __attribute__((ext_vector_type(16))) _Float16 v16h;
typedef __attribute__((ext_vector_type(8)))  _Float16 v8h;
typedef __attribute__((ext_vector_type(8)))  float    v8f;

#define NTOK 4096   // B*S
#define HD   512
#define SEQ  1024
#define HEADS 8
#define HDIM 64

// A-fragment (16x32 f16) per-lane data = two contiguous 16B runs:
// halves [hi*8 .. hi*8+7] and [hi*8+16 .. hi*8+23] of the logical 32-K row.
// `row` must already include the +hi*8 offset (16B aligned).
__device__ __forceinline__ v16h load_a_frag(const _Float16* row) {
  v8h lo = *(const v8h*)(row);
  v8h hc = *(const v8h*)(row + 16);
  return __builtin_shufflevector(lo, hc, 0, 1, 2, 3, 4, 5, 6, 7,
                                 8, 9, 10, 11, 12, 13, 14, 15);
}

// ================= Kernel A: p = layernorm(tanh(x @ Wp.T + bp)) ==========
__global__ void proj_ln_kernel(const float* __restrict__ x,
                               const float* __restrict__ Wp,
                               const float* __restrict__ bp,
                               const float* __restrict__ g,
                               const float* __restrict__ bb,
                               float* __restrict__ p) {
  int wave = (blockIdx.x * blockDim.x + threadIdx.x) >> 5;
  int lane = threadIdx.x & 31;
  if (wave >= NTOK) return;
  const float* xr = x + (size_t)wave * HD;
  float a0 = 0.f, a1 = 0.f, a2 = 0.f, a3 = 0.f;
  for (int c = lane; c < HD; c += 32) {
    float xv = xr[c];
    a0 += xv * Wp[c];
    a1 += xv * Wp[HD + c];
    a2 += xv * Wp[2 * HD + c];
    a3 += xv * Wp[3 * HD + c];
  }
  for (int off = 16; off; off >>= 1) {
    a0 += __shfl_xor(a0, off);
    a1 += __shfl_xor(a1, off);
    a2 += __shfl_xor(a2, off);
    a3 += __shfl_xor(a3, off);
  }
  if (lane == 0) {
    float v0 = tanhf(a0 + bp[0]);
    float v1 = tanhf(a1 + bp[1]);
    float v2 = tanhf(a2 + bp[2]);
    float v3 = tanhf(a3 + bp[3]);
    float mu = 0.25f * (v0 + v1 + v2 + v3);
    float d0 = v0 - mu, d1 = v1 - mu, d2 = v2 - mu, d3 = v3 - mu;
    float var = 0.25f * (d0 * d0 + d1 * d1 + d2 * d2 + d3 * d3);
    float inv = rsqrtf(var + 1e-5f);
    float* pr = p + (size_t)wave * 4;
    pr[0] = d0 * inv * g[0] + bb[0];
    pr[1] = d1 * inv * g[1] + bb[1];
    pr[2] = d2 * inv * g[2] + bb[2];
    pr[3] = d3 * inv * g[3] + bb[3];
  }
}

// ================= Kernel B: 4-qubit circuit per (token, {q,k,v}) ========
__device__ __forceinline__ void gate1(float2* ps, int m,
                                      float2 u00, float2 u01,
                                      float2 u10, float2 u11) {
#pragma unroll
  for (int i = 0; i < 16; ++i) {
    if (i & m) continue;
    float2 a = ps[i], b = ps[i | m];
    float2 na, nb;
    na.x = u00.x * a.x - u00.y * a.y + u01.x * b.x - u01.y * b.y;
    na.y = u00.x * a.y + u00.y * a.x + u01.x * b.y + u01.y * b.x;
    nb.x = u10.x * a.x - u10.y * a.y + u11.x * b.x - u11.y * b.y;
    nb.y = u10.x * a.y + u10.y * a.x + u11.x * b.y + u11.y * b.x;
    ps[i] = na;
    ps[i | m] = nb;
  }
}

__device__ __forceinline__ void rzg(float2* ps, int m, float t) {
  float c = cosf(0.5f * t), s = sinf(0.5f * t);
#pragma unroll
  for (int i = 0; i < 16; ++i) {
    float sg = (i & m) ? s : -s;
    float2 a = ps[i];
    ps[i] = make_float2(a.x * c - a.y * sg, a.y * c + a.x * sg);
  }
}

__device__ __forceinline__ void xg(float2* ps, int m) {
#pragma unroll
  for (int i = 0; i < 16; ++i) {
    if (i & m) continue;
    float2 t = ps[i];
    ps[i] = ps[i | m];
    ps[i | m] = t;
  }
}

__device__ __forceinline__ void crxg(float2* ps, int mc, int mt, float t) {
  float c = cosf(0.5f * t), s = sinf(0.5f * t);
#pragma unroll
  for (int i = 0; i < 16; ++i) {
    if (!(i & mc) || (i & mt)) continue;
    float2 a = ps[i], b = ps[i | mt];
    ps[i]      = make_float2(c * a.x + s * b.y, c * a.y - s * b.x);
    ps[i | mt] = make_float2(c * b.x + s * a.y, c * b.y - s * a.x);
  }
}

__device__ __forceinline__ void cnotg(float2* ps, int mc, int mt) {
#pragma unroll
  for (int i = 0; i < 16; ++i) {
    if (!(i & mc) || (i & mt)) continue;
    float2 t = ps[i];
    ps[i] = ps[i | mt];
    ps[i | mt] = t;
  }
}

__global__ void vqc_kernel(const float* __restrict__ p,
                           const float* qw, const float* qe, const float* qg, const float* qb_,
                           const float* kw, const float* ke, const float* kg, const float* kb_,
                           const float* vw, const float* ve, const float* vg, const float* vb_,
                           float* __restrict__ tq, float* __restrict__ tk,
                           float* __restrict__ tv) {
  int idx = blockIdx.x * blockDim.x + threadIdx.x;
  if (idx >= 3 * NTOK) return;
  int which = idx >> 12;   // NTOK == 4096
  int tok   = idx & (NTOK - 1);
  const float* w  = (which == 0) ? qw : (which == 1) ? kw : vw;
  const float* ew = (which == 0) ? qe : (which == 1) ? ke : ve;
  const float* gm = (which == 0) ? qg : (which == 1) ? kg : vg;
  const float* bt = (which == 0) ? qb_ : (which == 1) ? kb_ : vb_;
  float* outp = ((which == 0) ? tq : (which == 1) ? tk : tv) + (size_t)tok * 12;

  float2 ps[16];
#pragma unroll
  for (int i = 0; i < 16; ++i) ps[i] = make_float2(0.f, 0.f);
  ps[0].x = 1.f;

  const float rr = 0.7071067811865476f;
  const float* ang = p + (size_t)tok * 4;
  // AngleEmbedding RY
#pragma unroll
  for (int i = 0; i < 4; ++i) {
    float c = cosf(0.5f * ang[i]), s = sinf(0.5f * ang[i]);
    gate1(ps, 8 >> i, make_float2(c, 0), make_float2(-s, 0),
          make_float2(s, 0), make_float2(c, 0));
  }
#pragma unroll
  for (int L = 0; L < 3; ++L) {
    // RX,RY,RZ with same angle per wire
#pragma unroll
    for (int i = 0; i < 4; ++i) {
      float t = w[L * 4 + i];
      float c = cosf(0.5f * t), s = sinf(0.5f * t);
      int m = 8 >> i;
      gate1(ps, m, make_float2(c, 0), make_float2(0, -s),
            make_float2(0, -s), make_float2(c, 0));           // RX
      gate1(ps, m, make_float2(c, 0), make_float2(-s, 0),
            make_float2(s, 0), make_float2(c, 0));            // RY
      rzg(ps, m, t);                                          // RZ
    }
    // CRX entangling ring
#pragma unroll
    for (int i = 0; i < 4; ++i)
      crxg(ps, 8 >> i, 8 >> ((i + 1) & 3), ew[L * 4 + i]);
    // Grover diffusion
#pragma unroll
    for (int q = 0; q < 4; ++q)
      gate1(ps, 8 >> q, make_float2(rr, 0), make_float2(rr, 0),
            make_float2(rr, 0), make_float2(-rr, 0));
#pragma unroll
    for (int q = 0; q < 4; ++q) xg(ps, 8 >> q);
    gate1(ps, 1, make_float2(rr, 0), make_float2(rr, 0),
          make_float2(rr, 0), make_float2(-rr, 0));
    { float2 t = ps[14]; ps[14] = ps[15]; ps[15] = t; }   // CCCX
    gate1(ps, 1, make_float2(rr, 0), make_float2(rr, 0),
          make_float2(rr, 0), make_float2(-rr, 0));
#pragma unroll
    for (int q = 0; q < 4; ++q) xg(ps, 8 >> q);
#pragma unroll
    for (int q = 0; q < 4; ++q)
      gate1(ps, 8 >> q, make_float2(rr, 0), make_float2(rr, 0),
            make_float2(rr, 0), make_float2(-rr, 0));
    // QAOA cost + mixer
    float gv = gm[L];
#pragma unroll
    for (int i = 0; i < 3; ++i) {
      cnotg(ps, 8 >> i, 8 >> (i + 1));
      rzg(ps, 8 >> (i + 1), gv);
      cnotg(ps, 8 >> i, 8 >> (i + 1));
    }
    float bv = bt[L];
    float c = cosf(0.5f * bv), s = sinf(0.5f * bv);
#pragma unroll
    for (int q = 0; q < 4; ++q)
      gate1(ps, 8 >> q, make_float2(c, 0), make_float2(0, -s),
            make_float2(0, -s), make_float2(c, 0));
  }
  // expvals: Z, X, Y per qubit
  float meas[12];
#pragma unroll
  for (int q = 0; q < 4; ++q) {
    int m = 8 >> q;
    float z = 0.f, xv = 0.f, yv = 0.f;
#pragma unroll
    for (int i = 0; i < 16; ++i) {
      float pr = ps[i].x * ps[i].x + ps[i].y * ps[i].y;
      z += (i & m) ? -pr : pr;
    }
#pragma unroll
    for (int i = 0; i < 16; ++i) {
      if (i & m) continue;
      float2 a = ps[i], b = ps[i | m];
      xv += a.x * b.x + a.y * b.y;    // Re(conj(a)*b)
      yv += a.x * b.y - a.y * b.x;    // Im(conj(a)*b)
    }
    meas[q] = z;
    meas[4 + q] = 2.f * xv;
    meas[8 + q] = 2.f * yv;
  }
#pragma unroll
  for (int j = 0; j < 12; ++j) outp[j] = meas[j];
}

// ============ Kernel C: tq/tk/tv (4096,12) -> f16 Q,K (BH,S,64), Vt (BH,64,S)
__global__ void qkv_proj_kernel(const float* __restrict__ tq,
                                const float* __restrict__ tk,
                                const float* __restrict__ tv,
                                const float* __restrict__ Wq, const float* __restrict__ bq,
                                const float* __restrict__ Wk, const float* __restrict__ bk,
                                const float* __restrict__ Wv, const float* __restrict__ bv,
                                _Float16* __restrict__ qh,
                                _Float16* __restrict__ kh,
                                _Float16* __restrict__ vt) {
  int idx = blockIdx.x * blockDim.x + threadIdx.x;   // exactly 3*4096*512
  int which = idx / (NTOK * HD);
  int rem = idx - which * (NTOK * HD);
  int tok = rem >> 9;
  int col = rem & (HD - 1);
  const float* t  = (which == 0) ? tq : (which == 1) ? tk : tv;
  const float* W  = (which == 0) ? Wq : (which == 1) ? Wk : Wv;
  const float* bb = (which == 0) ? bq : (which == 1) ? bk : bv;
  float acc = bb[col];
  const float* tr = t + (size_t)tok * 12;
  const float* wr = W + (size_t)col * 12;
#pragma unroll
  for (int j = 0; j < 12; ++j) acc += tr[j] * wr[j];
  int b = tok >> 10, s = tok & (SEQ - 1), h = col >> 6, d = col & (HDIM - 1);
  int bh = b * HEADS + h;
  if (which == 0)
    qh[((size_t)bh * SEQ + s) * HDIM + d] = (_Float16)(acc * 0.125f); // fold 1/sqrt(64)
  else if (which == 1)
    kh[((size_t)bh * SEQ + s) * HDIM + d] = (_Float16)acc;
  else
    vt[((size_t)bh * HDIM + d) * SEQ + s] = (_Float16)acc;           // transposed
}

// ================= Kernel D: flash attention, WMMA f16 ====================
// one wave per (bh, 16-query tile); keys in blocks of 32.
// Row sums of P come from an extra WMMA against an all-ones B matrix
// (C-fragment element j <-> row j+8*hi), removing 32 serialized
// ds_bpermute reductions per key block.
__global__ void attn_kernel(const _Float16* __restrict__ qh,
                            const _Float16* __restrict__ kh,
                            const _Float16* __restrict__ vt,
                            _Float16* __restrict__ oh) {
  __shared__ __align__(32) _Float16 lds[4 * 16 * 32];
  int wv = threadIdx.x >> 5;
  int lane = threadIdx.x & 31;
  int wid = blockIdx.x * 4 + wv;   // 0..2047
  int bh = wid >> 6;               // 0..31
  int qt = wid & 63;               // 0..63
  int q0 = qt * 16;
  int hi = lane >> 4;
  int ln = lane & 15;
  _Float16* tile = lds + wv * (16 * 32);

  const _Float16* qb = qh + (size_t)bh * SEQ * HDIM;
  const _Float16* kb = kh + (size_t)bh * SEQ * HDIM;
  const _Float16* vb = vt + (size_t)bh * HDIM * SEQ;

  // constant all-ones B fragment for row-sum WMMA
  v16h ones;
#pragma unroll
  for (int e = 0; e < 16; ++e) ones[e] = (_Float16)1.0f;

  // A fragments of Q tile (16 rows x 64, two K=32 chunks): 2x b128 each
  v16h aq[2];
#pragma unroll
  for (int c = 0; c < 2; ++c)
    aq[c] = load_a_frag(qb + (size_t)(q0 + ln) * HDIM + c * 32 + hi * 8);

  v8f o[4] = {};                       // 16x64 f32 accumulator
  float mstat[8], lstat[8];
#pragma unroll
  for (int j = 0; j < 8; ++j) { mstat[j] = -1e30f; lstat[j] = 0.f; }

  for (int n0 = 0; n0 < SEQ; n0 += 32) {
    // speculative prefetch of next key block (global_prefetch_b8; no LOADcnt)
    {
      int np = (n0 + 32 < SEQ) ? n0 + 32 : 0;
      __builtin_prefetch(kb + (size_t)(np + ln + 16 * hi) * HDIM, 0, 3);
      __builtin_prefetch(vb + (size_t)(2 * lane) * SEQ + np, 0, 3);
    }
    v8f s1 = {}, s2 = {};
#pragma unroll
    for (int c = 0; c < 2; ++c) {
      v16h b1 = load_a_frag(kb + (size_t)(n0 + ln) * HDIM + c * 32 + hi * 8);
      s1 = __builtin_amdgcn_wmma_f32_16x16x32_f16(false, aq[c], false, b1,
                                                  (short)0, s1, false, false);
      v16h b2 = load_a_frag(kb + (size_t)(n0 + 16 + ln) * HDIM + c * 32 + hi * 8);
      s2 = __builtin_amdgcn_wmma_f32_16x16x32_f16(false, aq[c], false, b2,
                                                  (short)0, s2, false, false);
    }
    // row max: butterfly with step-outer loop so 8 independent bpermutes
    // per step can be clause'd behind a single s_wait_dscnt
    float mx[8];
#pragma unroll
    for (int j = 0; j < 8; ++j) mx[j] = fmaxf(s1[j], s2[j]);
#pragma unroll
    for (int off = 1; off < 16; off <<= 1) {
#pragma unroll
      for (int j = 0; j < 8; ++j)
        mx[j] = fmaxf(mx[j], __shfl_xor(mx[j], off));
    }
    float alpha[8];
#pragma unroll
    for (int j = 0; j < 8; ++j) {
      float mnew = fmaxf(mstat[j], mx[j]);
      alpha[j] = __expf(mstat[j] - mnew);
      mstat[j] = mnew;
      float p1 = __expf(s1[j] - mnew);
      float p2 = __expf(s2[j] - mnew);
      int m = j + 8 * hi;
      tile[m * 32 + ln] = (_Float16)p1;        // C-layout -> LDS row-major
      tile[m * 32 + 16 + ln] = (_Float16)p2;
#pragma unroll
      for (int f = 0; f < 4; ++f) o[f][j] *= alpha[j];
    }
    // re-read P as A-fragment (2x ds_load_b128); DS ops in-order within wave
    v16h ap = load_a_frag(tile + ln * 32 + hi * 8);
    // row sums of P via WMMA against ones
    v8f rs = {};
    rs = __builtin_amdgcn_wmma_f32_16x16x32_f16(false, ap, false, ones,
                                                (short)0, rs, false, false);
#pragma unroll
    for (int j = 0; j < 8; ++j) lstat[j] = lstat[j] * alpha[j] + rs[j];
#pragma unroll
    for (int f = 0; f < 4; ++f) {
      v16h bvf = load_a_frag(vb + (size_t)(f * 16 + ln) * SEQ + n0 + hi * 8);
      o[f] = __builtin_amdgcn_wmma_f32_16x16x32_f16(false, ap, false, bvf,
                                                    (short)0, o[f], false, false);
    }
  }

  // normalize + store f16 O in (B,S,512) layout
  int b = bh >> 3, h = bh & 7;
#pragma unroll
  for (int f = 0; f < 4; ++f) {
#pragma unroll
    for (int j = 0; j < 8; ++j) {
      int m = j + 8 * hi;
      float v = o[f][j] / lstat[j];
      size_t tok = (size_t)b * SEQ + q0 + m;
      oh[tok * HD + h * HDIM + f * 16 + ln] = (_Float16)v;
    }
  }
}

// ================= Kernel: f32 -> f16 convert (for Wo) ====================
__global__ void cvt_kernel(const float* __restrict__ src,
                           _Float16* __restrict__ dst, int n) {
  int i = blockIdx.x * blockDim.x + threadIdx.x;
  if (i < n) dst[i] = (_Float16)src[i];
}

// ================= Kernel E: out = O @ Wo.T + bo, WMMA f16 ================
__global__ void outproj_kernel(const _Float16* __restrict__ oh,
                               const _Float16* __restrict__ woh,
                               const float* __restrict__ bo,
                               float* __restrict__ out) {
  int wv = threadIdx.x >> 5;
  int lane = threadIdx.x & 31;
  int wid = blockIdx.x * 8 + wv;   // 0..8191
  int tm = wid >> 5;               // 0..255 token tiles
  int tn = wid & 31;               // 0..31  col tiles
  int t0 = tm * 16, c0 = tn * 16;
  int hi = lane >> 4, ln = lane & 15;
  v8f acc = {};
  for (int j0 = 0; j0 < HD; j0 += 32) {
    v16h a = load_a_frag(oh + (size_t)(t0 + ln) * HD + j0 + hi * 8);
    v16h bfr = load_a_frag(woh + (size_t)(c0 + ln) * HD + j0 + hi * 8);
    acc = __builtin_amdgcn_wmma_f32_16x16x32_f16(false, a, false, bfr,
                                                 (short)0, acc, false, false);
  }
  int col = c0 + ln;
  float bias = bo[col];
#pragma unroll
  for (int j = 0; j < 8; ++j) {
    int rowi = t0 + j + 8 * hi;
    out[(size_t)rowi * HD + col] = acc[j] + bias;
  }
}

// ================= launch =================
extern "C" void kernel_launch(void* const* d_in, const int* in_sizes, int n_in,
                              void* d_out, int out_size, void* d_ws, size_t ws_size,
                              hipStream_t stream) {
  const float* x    = (const float*)d_in[0];
  const float* Wp   = (const float*)d_in[1];
  const float* bp   = (const float*)d_in[2];
  const float* ln_g = (const float*)d_in[3];
  const float* ln_b = (const float*)d_in[4];
  const float* qw = (const float*)d_in[5];
  const float* qe = (const float*)d_in[6];
  const float* qg = (const float*)d_in[7];
  const float* qb = (const float*)d_in[8];
  const float* kw = (const float*)d_in[9];
  const float* ke = (const float*)d_in[10];
  const float* kg = (const float*)d_in[11];
  const float* kb = (const float*)d_in[12];
  const float* vw = (const float*)d_in[13];
  const float* ve = (const float*)d_in[14];
  const float* vg = (const float*)d_in[15];
  const float* vb = (const float*)d_in[16];
  const float* Wq = (const float*)d_in[17];
  const float* bq = (const float*)d_in[18];
  const float* Wk = (const float*)d_in[19];
  const float* bk = (const float*)d_in[20];
  const float* Wv = (const float*)d_in[21];
  const float* bv = (const float*)d_in[22];
  const float* Wo = (const float*)d_in[23];
  const float* bo = (const float*)d_in[24];
  float* out = (float*)d_out;

  // workspace carve-up (base assumed >=256B aligned; all chunks 32B aligned)
  float* p  = (float*)d_ws;                   // 4096*4
  float* tq = p + NTOK * 4;                   // 4096*12
  float* tk = tq + NTOK * 12;
  float* tv = tk + NTOK * 12;
  _Float16* qh  = (_Float16*)(tv + NTOK * 12);        // 32*1024*64 f16
  _Float16* kh  = qh + (size_t)32 * SEQ * HDIM;
  _Float16* vt  = kh + (size_t)32 * SEQ * HDIM;
  _Float16* oh  = vt + (size_t)32 * SEQ * HDIM;       // 4096*512 f16
  _Float16* woh = oh + (size_t)NTOK * HD;             // 512*512 f16

  proj_ln_kernel<<<NTOK / 8, 256, 0, stream>>>(x, Wp, bp, ln_g, ln_b, p);
  vqc_kernel<<<(3 * NTOK) / 256, 256, 0, stream>>>(p,
      qw, qe, qg, qb, kw, ke, kg, kb, vw, ve, vg, vb, tq, tk, tv);
  qkv_proj_kernel<<<(3 * NTOK * HD) / 256, 256, 0, stream>>>(tq, tk, tv,
      Wq, bq, Wk, bk, Wv, bv, qh, kh, vt);
  cvt_kernel<<<(HD * HD) / 256, 256, 0, stream>>>(Wo, woh, HD * HD);
  attn_kernel<<<512, 128, 0, stream>>>(qh, kh, vt, oh);
  outproj_kernel<<<1024, 256, 0, stream>>>(oh, woh, bo, out);
}
